// node_update_layer_26517128085908
// MI455X (gfx1250) — matrix operand order repaired and checked by reading, not compiled, not run on hardware
//
#include <hip/hip_runtime.h>

// ---- problem constants ----
#define B_   1024
#define K_   64
#define IN_  128
#define CH_  256
#define ROWS (B_ * K_)           // 65536
#define OUTW (IN_ + CH_)         // 384
#define FEATW 384                // 3*IN
#define FSTRIDE 392              // padded LDS stride (halfs); bank-shift 4/row, 16B-aligned rows
#define HSTRIDE 264              // padded LDS stride for hidden tile
#define MT   4                   // M-tiles (of 16 rows) per block
#define RT   (MT * 16)           // rows per block = 64

typedef __attribute__((ext_vector_type(16))) _Float16 v16h;
typedef __attribute__((ext_vector_type(8)))  _Float16 v8h;
typedef __attribute__((ext_vector_type(8)))  float    v8f;

// ---------------- stats: mean + top-2 + argmax over K per (b, d) -------------
__global__ __launch_bounds__(256) void stats_kernel(
    const float* __restrict__ user, float* __restrict__ mean,
    float* __restrict__ max1, float* __restrict__ max2, int* __restrict__ amax)
{
    int idx = blockIdx.x * 256 + threadIdx.x;          // B*IN = 131072
    int b = idx >> 7;                                  // / IN_
    int d = idx & 127;
    const float* p = user + (size_t)b * K_ * IN_ + d;
    float s = 0.f, m1 = -3.402823466e38f, m2 = -3.402823466e38f;
    int am = 0;
    #pragma unroll 4
    for (int k = 0; k < K_; ++k) {
        float v = p[(size_t)k * IN_];
        s += v;
        if (v > m1) { m2 = m1; m1 = v; am = k; }
        else if (v > m2) { m2 = v; }
    }
    mean[idx] = s * (1.f / K_);
    max1[idx] = m1;
    max2[idx] = m2;
    amax[idx] = am;
}

// ---- pack weights (Kdim x 256, row-major fp32) into f16 WMMA B-fragments ----
// layout: tile(kt,nt) -> 32 lanes x 16 halfs contiguous;
// B frag element (lane, i) = W[kt*32 + (i>>3)*16 + (lane>>4)*8 + (i&7)][nt*16 + (lane&15)]
__global__ __launch_bounds__(256) void pack_kernel(
    const float* __restrict__ W, _Float16* __restrict__ Wp, int n)
{
    int idx = blockIdx.x * 256 + threadIdx.x;
    if (idx >= n) return;
    int tile = idx >> 9;          // / 512
    int rem  = idx & 511;
    int lane = rem >> 4;
    int i    = rem & 15;
    int kt = tile >> 4;           // / 16
    int nt = tile & 15;
    int k = kt * 32 + (i >> 3) * 16 + (lane >> 4) * 8 + (i & 7);
    int ncol = nt * 16 + (lane & 15);
    Wp[idx] = (_Float16)W[(size_t)k * CH_ + ncol];
}

// ---- A-fragment gather from an LDS f16 tile (ISA 16-bit A 16x32 layout) ----
__device__ __forceinline__ v16h load_a(const _Float16* lds, int stride, int kt, int lane)
{
    int row = lane & 15, g = lane >> 4;
    const _Float16* base = lds + row * stride + kt * 32 + g * 8;
    union { v16h v; v8h h[2]; } u;
    u.h[0] = *(const v8h*)(base);        // K = kt*32 + g*8 + (0..7)
    u.h[1] = *(const v8h*)(base + 16);   // K = kt*32 + 16 + g*8 + (0..7)
    return u.v;
}

__device__ __forceinline__ v16h load_b(const _Float16* wp, int tile, int lane)
{
    return ((const v16h*)wp)[tile * 32 + lane];
}

__device__ __forceinline__ v8f wmma_f16(v16h a, v16h b, v8f c)
{
    return __builtin_amdgcn_wmma_f32_16x16x32_f16(false, a, false, b, (short)0, c, false, false);
}

// -------- main user-node MLP: 64 rows (4 M-tiles) per 256-thread block -------
__global__ __launch_bounds__(256) void user_kernel(
    const float* __restrict__ user, const float* __restrict__ ris,
    const float* __restrict__ b1v, const float* __restrict__ b2v,
    const _Float16* __restrict__ W1p, const _Float16* __restrict__ W2p,
    const float* __restrict__ max1, const float* __restrict__ max2,
    const int* __restrict__ amax, float* __restrict__ out)
{
    __shared__ __align__(16) _Float16 sfeat[RT * FSTRIDE];  // ~50 KB
    __shared__ __align__(16) _Float16 sh[RT * HSTRIDE];     // ~34 KB

    const int tid  = threadIdx.x;
    const int lane = tid & 31;
    const int wave = tid >> 5;
    const long rowbase = (long)blockIdx.x * RT;

    // build feat tile (f16) + fused passthrough copy of user_node
    for (int idx = tid; idx < RT * FEATW; idx += 256) {
        int r = idx / FEATW, c = idx - r * FEATW;
        long grow = rowbase + r;
        int b = (int)(grow >> 6), k = (int)(grow & 63);
        float v;
        if (c < IN_) {
            v = user[(grow << 7) + c];
            out[grow * OUTW + c] = v;                 // passthrough cols 0..127
        } else if (c < 2 * IN_) {
            int d = c - IN_;
            float m1 = max1[b * IN_ + d];
            float m2 = max2[b * IN_ + d];
            v = (amax[b * IN_ + d] == k) ? m2 : m1;   // exact leave-one-out max
        } else {
            v = ris[b * IN_ + (c - 2 * IN_)];
        }
        sfeat[r * FSTRIDE + c] = (_Float16)v;
    }
    __syncthreads();

    const int nt0 = wave * 2, nt1 = nt0 + 1;
    const int col0 = nt0 * 16 + (lane & 15), col1 = col0 + 16;
    const int mrow = (lane >> 4) * 8;

    v8f acc[MT][2];
    { float bv0 = b1v[col0], bv1 = b1v[col1];
      for (int m = 0; m < MT; ++m)
          for (int j = 0; j < 8; ++j) { acc[m][0][j] = bv0; acc[m][1][j] = bv1; } }

    for (int kt = 0; kt < 12; ++kt) {
        v16h w0 = load_b(W1p, kt * 16 + nt0, lane);
        v16h w1 = load_b(W1p, kt * 16 + nt1, lane);
        #pragma unroll
        for (int m = 0; m < MT; ++m) {
            v16h a = load_a(sfeat + m * 16 * FSTRIDE, FSTRIDE, kt, lane);
            acc[m][0] = wmma_f16(a, w0, acc[m][0]);
            acc[m][1] = wmma_f16(a, w1, acc[m][1]);
        }
    }

    // ReLU -> hidden tile in LDS (f16)
    for (int m = 0; m < MT; ++m)
        for (int j = 0; j < 8; ++j) {
            int row = m * 16 + mrow + j;
            float v0 = acc[m][0][j] > 0.f ? acc[m][0][j] : 0.f;
            float v1 = acc[m][1][j] > 0.f ? acc[m][1][j] : 0.f;
            sh[row * HSTRIDE + col0] = (_Float16)v0;
            sh[row * HSTRIDE + col1] = (_Float16)v1;
        }
    __syncthreads();

    { float bv0 = b2v[col0], bv1 = b2v[col1];
      for (int m = 0; m < MT; ++m)
          for (int j = 0; j < 8; ++j) { acc[m][0][j] = bv0; acc[m][1][j] = bv1; } }

    for (int kt = 0; kt < 8; ++kt) {
        v16h w0 = load_b(W2p, kt * 16 + nt0, lane);
        v16h w1 = load_b(W2p, kt * 16 + nt1, lane);
        #pragma unroll
        for (int m = 0; m < MT; ++m) {
            v16h a = load_a(sh + m * 16 * HSTRIDE, HSTRIDE, kt, lane);
            acc[m][0] = wmma_f16(a, w0, acc[m][0]);
            acc[m][1] = wmma_f16(a, w1, acc[m][1]);
        }
    }

    for (int m = 0; m < MT; ++m)
        for (int j = 0; j < 8; ++j) {
            long grow = rowbase + m * 16 + mrow + j;
            out[grow * OUTW + IN_ + col0] = acc[m][0][j];
            out[grow * OUTW + IN_ + col1] = acc[m][1][j];
        }
}

// ---------------- RIS MLP: rows = B, rfeat = [RIS, mean_user] ----------------
__global__ __launch_bounds__(256) void ris_kernel(
    const float* __restrict__ ris, const float* __restrict__ mean,
    const float* __restrict__ b3v, const float* __restrict__ b4v,
    const _Float16* __restrict__ W3p, const _Float16* __restrict__ W4p,
    float* __restrict__ outR)
{
    __shared__ __align__(16) _Float16 sfeat[16 * HSTRIDE];   // 256-wide feat
    __shared__ __align__(16) _Float16 sh[16 * HSTRIDE];

    const int tid  = threadIdx.x;
    const int lane = tid & 31;
    const int wave = tid >> 5;
    const int rowbase = blockIdx.x * 16;

    for (int idx = tid; idx < 16 * 256; idx += 256) {
        int r = idx >> 8, c = idx & 255;
        int b = rowbase + r;
        float v;
        if (c < IN_) {
            v = ris[b * IN_ + c];
            outR[(size_t)b * OUTW + c] = v;          // passthrough
        } else {
            v = mean[b * IN_ + (c - IN_)];
        }
        sfeat[r * HSTRIDE + c] = (_Float16)v;
    }
    __syncthreads();

    const int nt0 = wave * 2, nt1 = nt0 + 1;
    const int col0 = nt0 * 16 + (lane & 15), col1 = col0 + 16;
    const int mrow = (lane >> 4) * 8;

    v8f acc0, acc1;
    { float bv0 = b3v[col0], bv1 = b3v[col1];
      for (int j = 0; j < 8; ++j) { acc0[j] = bv0; acc1[j] = bv1; } }

    for (int kt = 0; kt < 8; ++kt) {
        v16h a  = load_a(sfeat, HSTRIDE, kt, lane);
        v16h w0 = load_b(W3p, kt * 16 + nt0, lane);
        v16h w1 = load_b(W3p, kt * 16 + nt1, lane);
        acc0 = wmma_f16(a, w0, acc0);
        acc1 = wmma_f16(a, w1, acc1);
    }
    for (int j = 0; j < 8; ++j) {
        float v0 = acc0[j] > 0.f ? acc0[j] : 0.f;
        float v1 = acc1[j] > 0.f ? acc1[j] : 0.f;
        sh[(mrow + j) * HSTRIDE + col0] = (_Float16)v0;
        sh[(mrow + j) * HSTRIDE + col1] = (_Float16)v1;
    }
    __syncthreads();

    { float bv0 = b4v[col0], bv1 = b4v[col1];
      for (int j = 0; j < 8; ++j) { acc0[j] = bv0; acc1[j] = bv1; } }

    for (int kt = 0; kt < 8; ++kt) {
        v16h a  = load_a(sh, HSTRIDE, kt, lane);
        v16h w0 = load_b(W4p, kt * 16 + nt0, lane);
        v16h w1 = load_b(W4p, kt * 16 + nt1, lane);
        acc0 = wmma_f16(a, w0, acc0);
        acc1 = wmma_f16(a, w1, acc1);
    }
    for (int j = 0; j < 8; ++j) {
        int b = rowbase + mrow + j;
        outR[(size_t)b * OUTW + IN_ + col0] = acc0[j];
        outR[(size_t)b * OUTW + IN_ + col1] = acc1[j];
    }
}

extern "C" void kernel_launch(void* const* d_in, const int* in_sizes, int n_in,
                              void* d_out, int out_size, void* d_ws, size_t ws_size,
                              hipStream_t stream)
{
    (void)in_sizes; (void)n_in; (void)out_size; (void)ws_size;
    const float* user = (const float*)d_in[0];
    const float* ris  = (const float*)d_in[1];
    // d_in[2] (cas_channel) and d_in[3] (d_link) are unused by the reference
    const float* W1 = (const float*)d_in[4];
    const float* b1 = (const float*)d_in[5];
    const float* W2 = (const float*)d_in[6];
    const float* b2 = (const float*)d_in[7];
    const float* W3 = (const float*)d_in[8];
    const float* b3 = (const float*)d_in[9];
    const float* W4 = (const float*)d_in[10];
    const float* b4 = (const float*)d_in[11];

    char* ws = (char*)d_ws;
    float*    mean = (float*)(ws + 0);          // B*IN f32      = 512 KB
    float*    max1 = (float*)(ws + 524288);     // B*IN f32
    float*    max2 = (float*)(ws + 1048576);    // B*IN f32
    int*      amax = (int*)  (ws + 1572864);    // B*IN i32
    _Float16* W1p  = (_Float16*)(ws + 2097152); // 384*256 f16 = 192 KB
    _Float16* W2p  = (_Float16*)(ws + 2293760); // 256*256 f16 = 128 KB
    _Float16* W3p  = (_Float16*)(ws + 2424832);
    _Float16* W4p  = (_Float16*)(ws + 2555904);

    float* out  = (float*)d_out;
    float* outR = out + (size_t)ROWS * OUTW;

    stats_kernel<<<(B_ * IN_) / 256, 256, 0, stream>>>(user, mean, max1, max2, amax);
    pack_kernel<<<(384 * 256) / 256, 256, 0, stream>>>(W1, W1p, 384 * 256);
    pack_kernel<<<(256 * 256) / 256, 256, 0, stream>>>(W2, W2p, 256 * 256);
    pack_kernel<<<(256 * 256) / 256, 256, 0, stream>>>(W3, W3p, 256 * 256);
    pack_kernel<<<(256 * 256) / 256, 256, 0, stream>>>(W4, W4p, 256 * 256);
    user_kernel<<<ROWS / RT, 256, 0, stream>>>(user, ris, b1, b2, W1p, W2p,
                                               max1, max2, amax, out);
    ris_kernel<<<B_ / 16, 256, 0, stream>>>(ris, mean, b3, b4, W3p, W4p, outR);
}
